// A3TGCN_56659208569225
// MI455X (gfx1250) — matrix operand order repaired and checked by reading, not compiled, not run on hardware
//
#include <hip/hip_runtime.h>

// ---------------- problem constants ----------------
#define NN   50000
#define FF   16
#define TT   12
#define HH   128
#define EE   800000
#define OUTD 16
#define FT   (FF*TT)   // 192 channels propagated through the graph

typedef __bf16 bf16;
typedef __attribute__((ext_vector_type(16))) __bf16 v16bf;
typedef __attribute__((ext_vector_type(8)))  float  v8f;

// fast transcendentals on the hardware v_exp_f32 / v_rcp_f32 path
__device__ __forceinline__ float fast_rcp(float x) { return __builtin_amdgcn_rcpf(x); }
__device__ __forceinline__ float fast_sigmoid(float x) {
  return fast_rcp(1.f + __expf(-x));
}
__device__ __forceinline__ float fast_tanh(float x) {
  return 1.f - 2.f * fast_rcp(1.f + __expf(2.f * x));
}

// ---------------- WMMA A-operand loader (wave32, 16x16x32 bf16) ----------------
// 16(M) x 32(K) tile from row-major [*, lda] bf16 matrix (global or LDS).
// Contiguous K order (matches packed B): lanes 0-15 row M=lane hold K=0..15;
// lanes 16-31 row M=lane-16 hold K=16..31.
__device__ __forceinline__ v16bf load_a16(const bf16* p, int lda) {
  int lane = threadIdx.x & 31;
  int m = lane & 15, half = lane >> 4;
  v16bf a;
#pragma unroll
  for (int j = 0; j < 16; ++j) {
    int k = 16 * half + j;
    a[j] = p[(size_t)m * lda + k];
  }
  return a;
}

// B operands are pre-packed to per-lane register layout: each lane reads its
// 16 bf16 (32 bytes) contiguously.
#define WMMA_BF16(acc, a, b) \
  __builtin_amdgcn_wmma_f32_16x16x32_bf16(false, (a), false, (b), (short)0, (acc), false, false)

// ---------------- graph phase (fp32, L2-resident) ----------------
__global__ void k_deg_init(float* deg) {
  int i = blockIdx.x * blockDim.x + threadIdx.x;
  if (i < NN) deg[i] = 1.0f;                    // self-loop weight
}

__global__ void k_deg_scatter(const int* __restrict__ dst, const float* __restrict__ w,
                              float* deg) {
  int e = blockIdx.x * blockDim.x + threadIdx.x;
  if (e < EE) atomicAdd(&deg[dst[e]], w[e]);
}

__global__ void k_dinv(const float* __restrict__ deg, float* dinv) {
  int i = blockIdx.x * blockDim.x + threadIdx.x;
  if (i < NN) { float d = deg[i]; dinv[i] = d > 0.f ? rsqrtf(d) : 0.f; }
}

// xprop initialized with the self-loop term: dinv_i^2 * x_i  (writes every element)
__global__ void k_selfinit(const float* __restrict__ x, const float* __restrict__ dinv,
                           float* xprop) {
  size_t idx = (size_t)blockIdx.x * blockDim.x + threadIdx.x;
  if (idx < (size_t)NN * FT) {
    int i = (int)(idx / FT);
    float d = dinv[i];
    xprop[idx] = d * d * x[idx];
  }
}

// one wave per edge, 192 channels striped over 32 lanes (6 atomics each)
__global__ void k_edge_scatter(const int* __restrict__ src, const int* __restrict__ dst,
                               const float* __restrict__ w, const float* __restrict__ dinv,
                               const float* __restrict__ x, float* xprop) {
  int e    = (int)(((size_t)blockIdx.x * blockDim.x + threadIdx.x) >> 5);
  int lane = threadIdx.x & 31;
  if (e >= EE) return;
  int s = src[e], d = dst[e];
  float nrm = dinv[s] * w[e] * dinv[d];
  const float* xs = x + (size_t)s * FT;
  float* xd = xprop + (size_t)d * FT;
#pragma unroll
  for (int c = lane; c < FT; c += 32)
    atomicAdd(&xd[c], nrm * xs[c]);
}

// xprop (N,16,12) fp32 -> xpb (N,12,32) bf16, features zero-padded 16->32 so the
// gate input term becomes one extra K=32 WMMA.
__global__ void k_xpb(const float* __restrict__ xprop, bf16* __restrict__ xpb) {
  size_t idx = (size_t)blockIdx.x * blockDim.x + threadIdx.x;
  if (idx >= (size_t)NN * TT * 32) return;
  int k = (int)(idx & 31);
  int t = (int)((idx >> 5) % TT);
  int n = (int)(idx / (32 * TT));
  xpb[idx] = (k < FF) ? (bf16)xprop[(size_t)n * FT + k * TT + t] : (bf16)0.f;
}

// ---------------- weight prep ----------------
// A_g = W_g @ L_g[:128]  (fp32 16x128 per gate)
__global__ void k_prep_A(const float* __restrict__ wz, const float* __restrict__ wr,
                         const float* __restrict__ wh,
                         const float* __restrict__ lz, const float* __restrict__ lr,
                         const float* __restrict__ lh, float* Atmp) {
  int idx = blockIdx.x * blockDim.x + threadIdx.x;
  if (idx >= 3 * FF * HH) return;
  int g = idx / (FF * HH), rem = idx % (FF * HH), f = rem / HH, j = rem % HH;
  const float* W = (g == 0) ? wz : (g == 1) ? wr : wh;
  const float* L = (g == 0) ? lz : (g == 1) ? lr : lh;
  float s = 0.f;
#pragma unroll 8
  for (int k = 0; k < HH; ++k) s += W[f * HH + k] * L[k * HH + j];
  Atmp[idx] = s;
}

// fused biases: c_g = b_g @ L_g[:128] + lin_b_g
__global__ void k_prep_c(const float* __restrict__ bz, const float* __restrict__ br,
                         const float* __restrict__ bh,
                         const float* __restrict__ lz, const float* __restrict__ lr,
                         const float* __restrict__ lh,
                         const float* __restrict__ lbz, const float* __restrict__ lbr,
                         const float* __restrict__ lbh, float* cc) {
  int idx = blockIdx.x * blockDim.x + threadIdx.x;
  if (idx >= 3 * HH) return;
  int g = idx / HH, j = idx % HH;
  const float* B  = (g == 0) ? bz  : (g == 1) ? br  : bh;
  const float* L  = (g == 0) ? lz  : (g == 1) ? lr  : lh;
  const float* LB = (g == 0) ? lbz : (g == 1) ? lbr : lbh;
  float s = LB[j];
#pragma unroll 8
  for (int k = 0; k < HH; ++k) s += B[k] * L[k * HH + j];
  cc[idx] = s;
}

// Pack all B operands into per-lane WMMA register layout.
// packed index r: j = r&15 (element), lane = (r>>4)&31, tile = r>>9 (= ctile*nkt+kk)
// source k = kk*32 + 16*(lane>>4) + j ; n = ctile*16 + (lane&15)
__global__ void k_pack(const float* __restrict__ lz, const float* __restrict__ lr,
                       const float* __restrict__ lh, const float* __restrict__ a1w,
                       const float* __restrict__ ow, const float* __restrict__ Atmp,
                       bf16* __restrict__ LbotP, bf16* __restrict__ att1P,
                       bf16* __restrict__ outwP, bf16* __restrict__ ApackP) {
  int idx = blockIdx.x * blockDim.x + threadIdx.x;
  const int SL = HH * HH;   // 16384
  if (idx < 3 * SL) {       // L_bot gates: K=128 (nkt=4), N=128
    int g = idx / SL, r = idx % SL;
    const float* L = (g == 0) ? lz : (g == 1) ? lr : lh;
    int j = r & 15, lane = (r >> 4) & 31, tile = r >> 9;
    int kk = tile & 3, ctile = tile >> 2;
    int k = kk * 32 + ((lane >> 4) << 4) + j;
    int n = ctile * 16 + (lane & 15);
    LbotP[idx] = (bf16)L[(size_t)(HH + k) * HH + n];
  } else if (idx < 4 * SL) {  // att1: K=128, N=128
    int r = idx - 3 * SL;
    int j = r & 15, lane = (r >> 4) & 31, tile = r >> 9;
    int kk = tile & 3, ctile = tile >> 2;
    int k = kk * 32 + ((lane >> 4) << 4) + j;
    int n = ctile * 16 + (lane & 15);
    att1P[r] = (bf16)a1w[(size_t)k * HH + n];
  } else if (idx < 4 * SL + HH * OUTD) {  // out_w: K=128 (nkt=4), N=16 (nct=1)
    int r = idx - 4 * SL;
    int j = r & 15, lane = (r >> 4) & 31, tile = r >> 9;   // tile == kk
    int k = tile * 32 + ((lane >> 4) << 4) + j;
    int n = lane & 15;
    outwP[r] = (bf16)ow[(size_t)k * OUTD + n];
  } else if (idx < 4 * SL + HH * OUTD + 3 * 32 * HH) {  // A gates: K=32 (zero-pad), N=128
    int r = idx - 4 * SL - HH * OUTD;
    int g = r / (32 * HH), rr = r % (32 * HH);
    int j = rr & 15, lane = (rr >> 4) & 31, ctile = rr >> 9;   // nkt=1
    int k = ((lane >> 4) << 4) + j;
    int n = ctile * 16 + (lane & 15);
    ApackP[r] = (k < FF) ? (bf16)Atmp[(size_t)(g * FF + k) * HH + n] : (bf16)0.f;
  }
}

__global__ void k_h_init(bf16* h) {
  size_t i = (size_t)blockIdx.x * blockDim.x + threadIdx.x;
  if (i < (size_t)NN * HH) h[i] = (bf16)0.f;
}

// ---------------- GRU timestep: z & r (16 waves: 8 z-tiles + 8 r-tiles) ----------------
__global__ void k_gru_zr(const bf16* __restrict__ h, const bf16* __restrict__ LbotP,
                         const bf16* __restrict__ ApackP, const float* __restrict__ cc,
                         const bf16* __restrict__ xpb,
                         bf16* __restrict__ zbuf, bf16* __restrict__ hrbuf, int t) {
  __shared__ bf16 hsm[16 * HH];    // 16 rows of h  (contiguous 4KB in global)
  __shared__ bf16 xsm[16 * 32];    // 16 rows of xpb at timestep t
  int tid = threadIdx.x, row0 = blockIdx.x * 16;
  *(unsigned long long*)(hsm + tid * 4) =
      *(const unsigned long long*)(h + (size_t)row0 * HH + tid * 4);     // 512*4 = 2048
  if (tid < 128) {
    int r = tid >> 3, k4 = (tid & 7) * 4;
    *(unsigned long long*)(xsm + r * 32 + k4) =
        *(const unsigned long long*)(xpb + ((size_t)(row0 + r) * TT + t) * 32 + k4);
  }
  __syncthreads();
  int wavei = tid >> 5, gate = wavei >> 3, ctile = wavei & 7;
  int lane = tid & 31, hi = lane >> 4, nl = lane & 15, ngl = ctile * 16 + nl;
  const bf16* Lp = LbotP + (size_t)gate * HH * HH;
  v8f acc = {};
#pragma unroll
  for (int kk = 0; kk < 4; ++kk) {
    v16bf a = load_a16(hsm + kk * 32, HH);
    v16bf b = *(const v16bf*)(Lp + (((size_t)ctile * 4 + kk) * 32 + lane) * 16);
    acc = WMMA_BF16(acc, a, b);
  }
  {  // gate input term as 5th WMMA (K=32, zero-padded)
    v16bf a = load_a16(xsm, 32);
    v16bf b = *(const v16bf*)(ApackP + (size_t)gate * 32 * HH + ((size_t)ctile * 32 + lane) * 16);
    acc = WMMA_BF16(acc, a, b);
  }
  float bias = cc[gate * HH + ngl];
#pragma unroll
  for (int v = 0; v < 8; ++v) {
    int m = v + 8 * hi;
    float s = fast_sigmoid(acc[v] + bias);
    size_t idx = (size_t)(row0 + m) * HH + ngl;
    if (gate == 0) zbuf[idx] = (bf16)s;
    else           hrbuf[idx] = (bf16)(s * (float)hsm[m * HH + ngl]);
  }
}

// ---------------- GRU timestep: h_tilde + state update + hs store ----------------
__global__ void k_gru_ht(bf16* __restrict__ h, const bf16* __restrict__ LbotP,
                         const bf16* __restrict__ ApackP, const float* __restrict__ cc,
                         const bf16* __restrict__ xpb, const bf16* __restrict__ zbuf,
                         const bf16* __restrict__ hrbuf, bf16* __restrict__ hs, int t) {
  __shared__ bf16 am [16 * HH];    // hr tile (A operand)
  __shared__ bf16 hsm[16 * HH];    // h tile
  __shared__ bf16 zsm[16 * HH];    // z tile
  __shared__ bf16 xsm[16 * 32];
  int tid = threadIdx.x, row0 = blockIdx.x * 16;
  ((float4*)am )[tid] = ((const float4*)(hrbuf + (size_t)row0 * HH))[tid];  // 256*8 = 2048
  ((float4*)hsm)[tid] = ((const float4*)(h     + (size_t)row0 * HH))[tid];
  ((float4*)zsm)[tid] = ((const float4*)(zbuf  + (size_t)row0 * HH))[tid];
  if (tid < 128) {
    int r = tid >> 3, k4 = (tid & 7) * 4;
    *(unsigned long long*)(xsm + r * 32 + k4) =
        *(const unsigned long long*)(xpb + ((size_t)(row0 + r) * TT + t) * 32 + k4);
  }
  __syncthreads();
  int ctile = tid >> 5, lane = tid & 31, hi = lane >> 4, nl = lane & 15, ngl = ctile * 16 + nl;
  const bf16* Lp = LbotP + (size_t)2 * HH * HH;
  v8f acc = {};
#pragma unroll
  for (int kk = 0; kk < 4; ++kk) {
    v16bf a = load_a16(am + kk * 32, HH);
    v16bf b = *(const v16bf*)(Lp + (((size_t)ctile * 4 + kk) * 32 + lane) * 16);
    acc = WMMA_BF16(acc, a, b);
  }
  {
    v16bf a = load_a16(xsm, 32);
    v16bf b = *(const v16bf*)(ApackP + (size_t)2 * 32 * HH + ((size_t)ctile * 32 + lane) * 16);
    acc = WMMA_BF16(acc, a, b);
  }
  float bias = cc[2 * HH + ngl];
#pragma unroll
  for (int v = 0; v < 8; ++v) {
    int m = v + 8 * hi;
    float htl = fast_tanh(acc[v] + bias);
    float z  = (float)zsm[m * HH + ngl];
    float ho = (float)hsm[m * HH + ngl];
    float hn = z * ho + (1.f - z) * htl;
    size_t idx = (size_t)(row0 + m) * HH + ngl;
    h[idx] = (bf16)hn;
    hs[((size_t)(row0 + m) * TT + t) * HH + ngl] = (bf16)hn;
  }
}

// ---------------- attention scores: relu(hs@att1+b1)·att2 + b2 ----------------
__global__ void k_att(const bf16* __restrict__ hs, const bf16* __restrict__ att1P,
                      const float* __restrict__ a1b, const float* __restrict__ a2w,
                      const float* __restrict__ a2b, float* __restrict__ scores) {
  __shared__ bf16 am[16 * HH];
  __shared__ float srow[16];
  int tid = threadIdx.x, row0 = blockIdx.x * 16;
  ((float4*)am)[tid] = ((const float4*)(hs + (size_t)row0 * HH))[tid];   // 4KB contiguous
  if (tid < 16) srow[tid] = 0.f;
  __syncthreads();
  int ctile = tid >> 5, lane = tid & 31, hi = lane >> 4, nl = lane & 15, ngl = ctile * 16 + nl;
  v8f acc = {};
#pragma unroll
  for (int kk = 0; kk < 4; ++kk) {
    v16bf a = load_a16(am + kk * 32, HH);
    v16bf b = *(const v16bf*)(att1P + (((size_t)ctile * 4 + kk) * 32 + lane) * 16);
    acc = WMMA_BF16(acc, a, b);
  }
  float w2 = a2w[ngl], b1 = a1b[ngl];
#pragma unroll
  for (int v = 0; v < 8; ++v) {
    float r = acc[v] + b1;
    r = r > 0.f ? r : 0.f;
    atomicAdd(&srow[v + 8 * hi], r * w2);
  }
  __syncthreads();
  if (tid < 16) scores[row0 + tid] = srow[tid] + a2b[0];
}

// ---------------- softmax over T + weighted pooling ----------------
__global__ void k_pool(const float* __restrict__ scores, const bf16* __restrict__ hs,
                       bf16* __restrict__ pooled) {
  int n = blockIdx.x, j = threadIdx.x;
  float s[TT], mx = -3.4e38f;
#pragma unroll
  for (int t = 0; t < TT; ++t) { s[t] = scores[n * TT + t]; mx = fmaxf(mx, s[t]); }
  float sum = 0.f;
#pragma unroll
  for (int t = 0; t < TT; ++t) { s[t] = __expf(s[t] - mx); sum += s[t]; }
  float inv = fast_rcp(sum), acc = 0.f;
#pragma unroll
  for (int t = 0; t < TT; ++t)
    acc += s[t] * inv * (float)hs[((size_t)n * TT + t) * HH + j];
  pooled[(size_t)n * HH + j] = (bf16)acc;
}

// ---------------- output projection: pooled(128) @ out_w(128x16) + b ----------------
__global__ void k_out(const bf16* __restrict__ pooled, const bf16* __restrict__ outwP,
                      const float* __restrict__ outb, float* __restrict__ out) {
  int wavei = threadIdx.x >> 5;
  int tile = blockIdx.x * 8 + wavei;
  if (tile >= NN / 16) return;
  int lane = threadIdx.x & 31, hi = lane >> 4, nl = lane & 15;
  v8f acc = {};
#pragma unroll
  for (int kk = 0; kk < 4; ++kk) {
    v16bf a = load_a16(pooled + (size_t)tile * 16 * HH + kk * 32, HH);
    v16bf b = *(const v16bf*)(outwP + ((size_t)kk * 32 + lane) * 16);
    acc = WMMA_BF16(acc, a, b);
  }
  float bb = outb[nl];
#pragma unroll
  for (int v = 0; v < 8; ++v) {
    int m = v + 8 * hi;
    out[(size_t)(tile * 16 + m) * OUTD + nl] = acc[v] + bb;
  }
}

// ---------------- host launch ----------------
extern "C" void kernel_launch(void* const* d_in, const int* in_sizes, int n_in,
                              void* d_out, int out_size, void* d_ws, size_t ws_size,
                              hipStream_t stream) {
  const float* x    = (const float*)d_in[0];
  const int*   eidx = (const int*)d_in[1];     // (2,E): row0=src, row1=dst
  const float* ew   = (const float*)d_in[2];
  const float* wz = (const float*)d_in[3],  *bz = (const float*)d_in[4];
  const float* wr = (const float*)d_in[5],  *br = (const float*)d_in[6];
  const float* wh = (const float*)d_in[7],  *bh = (const float*)d_in[8];
  const float* lz = (const float*)d_in[9],  *lbz = (const float*)d_in[10];
  const float* lr = (const float*)d_in[11], *lbr = (const float*)d_in[12];
  const float* lh = (const float*)d_in[13], *lbh = (const float*)d_in[14];
  const float* a1w = (const float*)d_in[15], *a1b = (const float*)d_in[16];
  const float* a2w = (const float*)d_in[17], *a2b = (const float*)d_in[18];
  const float* ow  = (const float*)d_in[19], *ob  = (const float*)d_in[20];
  float* out = (float*)d_out;
  const int* esrc = eidx;
  const int* edst = eidx + EE;

  // carve workspace
  char* p = (char*)d_ws;
  auto carve = [&](size_t bytes) -> char* {
    char* r = p; p += (bytes + 255) & ~(size_t)255; return r;
  };
  float* deg    = (float*)carve((size_t)NN * 4);
  float* dinv   = (float*)carve((size_t)NN * 4);
  float* xprop  = (float*)carve((size_t)NN * FT * 4);
  bf16*  xpb    = (bf16*) carve((size_t)NN * TT * 32 * 2);
  bf16*  h      = (bf16*) carve((size_t)NN * HH * 2);
  bf16*  zbuf   = (bf16*) carve((size_t)NN * HH * 2);
  bf16*  hrbuf  = (bf16*) carve((size_t)NN * HH * 2);
  bf16*  hs     = (bf16*) carve((size_t)NN * TT * HH * 2);
  float* scores = (float*)carve((size_t)NN * TT * 4);
  bf16*  pooled = (bf16*) carve((size_t)NN * HH * 2);
  float* Atmp   = (float*)carve((size_t)3 * FF * HH * 4);
  float* cc     = (float*)carve((size_t)3 * HH * 4);
  bf16*  LbotP  = (bf16*) carve((size_t)3 * HH * HH * 2);
  bf16*  att1P  = (bf16*) carve((size_t)HH * HH * 2);
  bf16*  outwP  = (bf16*) carve((size_t)HH * OUTD * 2);
  bf16*  ApackP = (bf16*) carve((size_t)3 * 32 * HH * 2);
  (void)ws_size; (void)in_sizes; (void)n_in; (void)out_size;

  // graph phase
  k_deg_init<<<(NN + 255) / 256, 256, 0, stream>>>(deg);
  k_deg_scatter<<<(EE + 255) / 256, 256, 0, stream>>>(edst, ew, deg);
  k_dinv<<<(NN + 255) / 256, 256, 0, stream>>>(deg, dinv);
  k_selfinit<<<(int)(((size_t)NN * FT + 255) / 256), 256, 0, stream>>>(x, dinv, xprop);
  k_edge_scatter<<<(EE * 32 + 255) / 256, 256, 0, stream>>>(esrc, edst, ew, dinv, x, xprop);
  k_xpb<<<(int)(((size_t)NN * TT * 32 + 255) / 256), 256, 0, stream>>>(xprop, xpb);

  // weight prep
  k_prep_A<<<(3 * FF * HH + 255) / 256, 256, 0, stream>>>(wz, wr, wh, lz, lr, lh, Atmp);
  k_prep_c<<<(3 * HH + 255) / 256, 256, 0, stream>>>(bz, br, bh, lz, lr, lh, lbz, lbr, lbh, cc);
  {
    int tot = 4 * HH * HH + HH * OUTD + 3 * 32 * HH;
    k_pack<<<(tot + 255) / 256, 256, 0, stream>>>(lz, lr, lh, a1w, ow, Atmp,
                                                  LbotP, att1P, outwP, ApackP);
  }
  k_h_init<<<(int)(((size_t)NN * HH + 255) / 256), 256, 0, stream>>>(h);

  // GRU recurrence (12 timesteps)
  for (int t = 0; t < TT; ++t) {
    k_gru_zr<<<NN / 16, 512, 0, stream>>>(h, LbotP, ApackP, cc, xpb, zbuf, hrbuf, t);
    k_gru_ht<<<NN / 16, 256, 0, stream>>>(h, LbotP, ApackP, cc, xpb, zbuf, hrbuf, hs, t);
  }

  // attention + pooling + output
  k_att<<<NN * TT / 16, 256, 0, stream>>>(hs, att1P, a1b, a2w, a2b, scores);
  k_pool<<<NN, HH, 0, stream>>>(scores, hs, pooled);
  k_out<<<(NN / 16 + 7) / 8, 256, 0, stream>>>(pooled, outwP, ob, out);
}
